// SpatioTemporalAttnBlock_52140902973458
// MI455X (gfx1250) — compile-verified, matrix operand order checked by reading, not run on hardware
//
#include <hip/hip_runtime.h>
#include <math.h>

// ---------------- CDNA5 WMMA plumbing (wave32, 16x16x32 bf16) ----------------
typedef __attribute__((ext_vector_type(16))) __bf16 v16bf;
typedef __attribute__((ext_vector_type(8)))  __bf16 v8bf;
typedef __attribute__((ext_vector_type(8)))  float  v8f;

#define DEVINL __device__ __forceinline__

// A/B fragment loader for 16-bit 16x32 (MxK) tiles, row-major source.
// Per ISA: lane L (L<16) holds row m=L, K = {0..7, 16..23}; lanes 16..31 hold
// K = {8..15, 24..31}. Caller passes p = base + row*ld + kb (kb = 0 or 8),
// so we read two contiguous 16B chunks at p and p+16.
DEVINL v16bf ldfrag(const __bf16* p) {
  v8bf lo = *(const v8bf*)(p);
  v8bf hi = *(const v8bf*)(p + 16);
  return __builtin_shufflevector(lo, hi, 0,1,2,3,4,5,6,7,8,9,10,11,12,13,14,15);
}

DEVINL v8f wmma_bf16(v16bf a, v16bf b, v8f c) {
  // (neg_a, A, neg_b, B, c_mod, C, reuse_a, reuse_b)
  return __builtin_amdgcn_wmma_f32_16x16x32_bf16(false, a, false, b, (short)0, c,
                                                 false, false);
}

// ---------------- problem dims ----------------
#define BB   4
#define CC   256
#define DDm  8
#define HHm  32
#define WWm  32
#define NNtok 8192          // H*W*D
#define NHH  4
#define DHH  64
#define PPm  64
#define MTOT (BB * NNtok)   // 32768 token rows

// =====================================================================
// K1: pack transposed weights to bf16 + zero reduction scratch
// =====================================================================
__global__ __launch_bounds__(256)
void pack_weights_kernel(const float* __restrict__ w_qkv,   // [256,768]
                         const float* __restrict__ w_out,   // [256,256]
                         const float* __restrict__ w_seq,   // [8192,64]
                         const float* __restrict__ pw_k,    // [256,256]
                         __bf16* __restrict__ wqkvT,        // [768,256]
                         __bf16* __restrict__ woutT,        // [256,256]
                         __bf16* __restrict__ wseqT,        // [64,8192]
                         __bf16* __restrict__ pwT,          // [256,256]
                         float* __restrict__ norms,         // [B,512]
                         float* __restrict__ kvacc)         // 131072 f32
{
  int idx = blockIdx.x * 256 + threadIdx.x;
  if (idx < 768 * 256) {
    int j = idx >> 8, i = idx & 255;
    wqkvT[idx] = (__bf16)w_qkv[i * 768 + j];
  } else if (idx < 768 * 256 + 256 * 256) {
    int o = idx - 768 * 256;
    int j = o >> 8, i = o & 255;
    woutT[o] = (__bf16)w_out[i * 256 + j];
  } else if (idx < 768 * 256 + 256 * 256 + 64 * 8192) {
    int o = idx - (768 * 256 + 256 * 256);
    int p = o >> 13, n = o & 8191;
    wseqT[o] = (__bf16)w_seq[n * 64 + p];
  } else if (idx < 768 * 256 + 256 * 256 + 64 * 8192 + 256 * 256) {
    int o = idx - (768 * 256 + 256 * 256 + 64 * 8192);
    pwT[o] = (__bf16)pw_k[o];             // pw_k already [cout][cin]
  } else if (idx < 768 * 256 + 256 * 256 + 64 * 8192 + 256 * 256 + 2048) {
    norms[idx - (768 * 256 + 256 * 256 + 64 * 8192 + 256 * 256)] = 0.f;
  } else if (idx < 768 * 256 + 256 * 256 + 64 * 8192 + 256 * 256 + 2048 + 131072) {
    kvacc[idx - (768 * 256 + 256 * 256 + 64 * 8192 + 256 * 256 + 2048)] = 0.f;
  }
}

// =====================================================================
// K2: x [B,C,D,H,W] f32 -> t [B,N,C] bf16 ; n = h*W*D + w*D + d
// =====================================================================
__global__ __launch_bounds__(256)
void pack_x_kernel(const float* __restrict__ x, __bf16* __restrict__ t) {
  int idx = blockIdx.x * 256 + threadIdx.x;          // b,n,c with c fastest
  int c = idx & 255;
  int n = (idx >> 8) & (NNtok - 1);
  int b = idx >> 21;
  int d = n & 7, w = (n >> 3) & 31, h = n >> 8;
  t[idx] = (__bf16)x[(((size_t)(b * CC + c) * DDm + d) * HHm + h) * WWm + w];
}

// =====================================================================
// K3: generic bf16 WMMA GEMM.  C_f32[M,Nn] = A[M,K] * (BT[Nn,K])^T + bias
// Wave computes a 32x64 tile (2 M-subtiles x 4 N-subtiles, 8 accumulators).
// K-loop is register double-buffered: WMMAs consume fragments loaded on the
// previous iteration so loads overlap compute (partial s_wait_loadcnt).
// =====================================================================
__global__ __launch_bounds__(256)
void gemm_bf16_kernel(const __bf16* __restrict__ A, const __bf16* __restrict__ BT,
                      const float* __restrict__ bias, float* __restrict__ Cf,
                      int M, int Nn, int K) {
  int wave = (blockIdx.x * 256 + threadIdx.x) >> 5;
  int ntiles = Nn >> 6;
  if (wave >= (M >> 5) * ntiles) return;
  int tm = wave / ntiles, tn = wave % ntiles;
  int lane = threadIdx.x & 31;
  int row = lane & 15;
  int kb = (lane >> 4) << 3;
  int mh = kb;

  const __bf16* Ap0 = A + (size_t)(tm * 32 + row) * K + kb;
  const __bf16* Ap1 = Ap0 + (size_t)16 * K;
  const __bf16* Bp0 = BT + (size_t)(tn * 64 + row) * K + kb;
  const size_t bstep = (size_t)16 * K;

  v8f acc0[4] = {v8f{}, v8f{}, v8f{}, v8f{}};
  v8f acc1[4] = {v8f{}, v8f{}, v8f{}, v8f{}};

  // prologue: fragments for k = 0
  v16bf a0 = ldfrag(Ap0), a1 = ldfrag(Ap1);
  v16bf b0 = ldfrag(Bp0);
  v16bf b1 = ldfrag(Bp0 + bstep);
  v16bf b2 = ldfrag(Bp0 + 2 * bstep);
  v16bf b3 = ldfrag(Bp0 + 3 * bstep);

  for (int k0 = 32; k0 < K; k0 += 32) {
    // issue next-iteration loads first (overlap with WMMAs below)
    v16bf na0 = ldfrag(Ap0 + k0), na1 = ldfrag(Ap1 + k0);
    v16bf nb0 = ldfrag(Bp0 + k0);
    v16bf nb1 = ldfrag(Bp0 + bstep + k0);
    v16bf nb2 = ldfrag(Bp0 + 2 * bstep + k0);
    v16bf nb3 = ldfrag(Bp0 + 3 * bstep + k0);
    __builtin_prefetch(Ap0 + k0 + 64, 0, 1);   // global_prefetch, no counter

    acc0[0] = wmma_bf16(a0, b0, acc0[0]);
    acc1[0] = wmma_bf16(a1, b0, acc1[0]);
    acc0[1] = wmma_bf16(a0, b1, acc0[1]);
    acc1[1] = wmma_bf16(a1, b1, acc1[1]);
    acc0[2] = wmma_bf16(a0, b2, acc0[2]);
    acc1[2] = wmma_bf16(a1, b2, acc1[2]);
    acc0[3] = wmma_bf16(a0, b3, acc0[3]);
    acc1[3] = wmma_bf16(a1, b3, acc1[3]);

    a0 = na0; a1 = na1; b0 = nb0; b1 = nb1; b2 = nb2; b3 = nb3;
  }
  // epilogue: last K-step
  acc0[0] = wmma_bf16(a0, b0, acc0[0]);
  acc1[0] = wmma_bf16(a1, b0, acc1[0]);
  acc0[1] = wmma_bf16(a0, b1, acc0[1]);
  acc1[1] = wmma_bf16(a1, b1, acc1[1]);
  acc0[2] = wmma_bf16(a0, b2, acc0[2]);
  acc1[2] = wmma_bf16(a1, b2, acc1[2]);
  acc0[3] = wmma_bf16(a0, b3, acc0[3]);
  acc1[3] = wmma_bf16(a1, b3, acc1[3]);

#pragma unroll
  for (int j = 0; j < 4; ++j) {
    int col = tn * 64 + j * 16 + row;
    float bj = bias ? bias[col] : 0.f;
#pragma unroll
    for (int r = 0; r < 8; ++r) {
      Cf[(size_t)(tm * 32 + mh + r) * Nn + col]      = acc0[j][r] + bj;
      Cf[(size_t)(tm * 32 + 16 + mh + r) * Nn + col] = acc1[j][r] + bj;
    }
  }
}

// =====================================================================
// K4: per-(b, feature) sum of squares over tokens for q,k (cols 0..511)
// =====================================================================
__global__ __launch_bounds__(256)
void colnorm_kernel(const float* __restrict__ qkv, float* __restrict__ norms) {
  int b = blockIdx.x >> 7;
  int n0 = (blockIdx.x & 127) * 64;
#pragma unroll
  for (int half = 0; half < 2; ++half) {
    int c = threadIdx.x + half * 256;      // 0..511
    float s = 0.f;
    for (int r = 0; r < 64; ++r) {
      float v = qkv[(size_t)(b * NNtok + n0 + r) * 768 + c];
      s += v * v;
    }
    atomicAdd(norms + b * 512 + c, s);
  }
}

__global__ __launch_bounds__(256)
void norm_finish_kernel(float* __restrict__ norms) {
  int i = blockIdx.x * 256 + threadIdx.x;
  if (i < BB * 512) {
    float s = sqrtf(norms[i]);
    norms[i] = 1.f / fmaxf(s, 1e-12f);
  }
}

// =====================================================================
// K5: split qkv -> normalized q bf16 [bh,N,dh]; kT,vT bf16 [bh,dh,N]
// =====================================================================
__global__ __launch_bounds__(256)
void pack_qkv_kernel(const float* __restrict__ qkv, const float* __restrict__ inv,
                     __bf16* __restrict__ q, __bf16* __restrict__ kT,
                     __bf16* __restrict__ vT) {
  int idx = blockIdx.x * 256 + threadIdx.x;
  int c = idx & 255;
  int n = (idx >> 8) & (NNtok - 1);
  int b = idx >> 21;
  int h = c >> 6, cf = c & 63;
  int bh = b * NHH + h;
  size_t row = (size_t)(b * NNtok + n) * 768;
  float qv = qkv[row + c] * inv[b * 512 + c];
  float kv = qkv[row + 256 + c] * inv[b * 512 + 256 + c];
  float vv = qkv[row + 512 + c];
  q[((size_t)bh * NNtok + n) * DHH + cf] = (__bf16)qv;
  kT[((size_t)bh * DHH + cf) * NNtok + n] = (__bf16)kv;
  vT[((size_t)bh * DHH + cf) * NNtok + n] = (__bf16)vv;
}

// =====================================================================
// K6: split-K WMMA GEMMs for sequence projections.
//   which=0: kpT[p][dh]  = wseqT[64,8192] x kT[bh]       (K = 8192)
//   which=1: vp [dh][p]  = vT[bh][64,8192] x wseqT
// grid = B*NH * 32 chunks * 2 ; block = 4 waves, each a 16-row strip.
// K-loop double-buffered like the main GEMM.
// =====================================================================
__global__ __launch_bounds__(128)
void kvproj_kernel(const __bf16* __restrict__ wseqT, const __bf16* __restrict__ kT,
                   const __bf16* __restrict__ vT, float* __restrict__ acc) {
  int id = blockIdx.x;
  int which = id & 1;
  int ck = (id >> 1) & 31;          // 32 K-chunks of 256
  int bh = id >> 6;
  int wv = threadIdx.x >> 5, lane = threadIdx.x & 31;
  int row = lane & 15;
  int kb = (lane >> 4) << 3;
  int mh = kb;
  int k0base = ck * 256;

  const __bf16* Abase = (which == 0) ? wseqT : (vT + (size_t)bh * DHH * NNtok);
  const __bf16* Bbase = (which == 0) ? (kT + (size_t)bh * DHH * NNtok) : wseqT;

  const __bf16* Ap = Abase + (size_t)(wv * 16 + row) * NNtok + k0base + kb;
  const __bf16* Bp = Bbase + (size_t)row * NNtok + k0base + kb;
  const size_t bstep = (size_t)16 * NNtok;

  v8f a4[4] = {v8f{}, v8f{}, v8f{}, v8f{}};
  v16bf a = ldfrag(Ap);
  v16bf b0 = ldfrag(Bp);
  v16bf b1 = ldfrag(Bp + bstep);
  v16bf b2 = ldfrag(Bp + 2 * bstep);
  v16bf b3 = ldfrag(Bp + 3 * bstep);
  for (int k = 32; k < 256; k += 32) {
    v16bf na = ldfrag(Ap + k);
    v16bf nb0 = ldfrag(Bp + k);
    v16bf nb1 = ldfrag(Bp + bstep + k);
    v16bf nb2 = ldfrag(Bp + 2 * bstep + k);
    v16bf nb3 = ldfrag(Bp + 3 * bstep + k);
    a4[0] = wmma_bf16(a, b0, a4[0]);
    a4[1] = wmma_bf16(a, b1, a4[1]);
    a4[2] = wmma_bf16(a, b2, a4[2]);
    a4[3] = wmma_bf16(a, b3, a4[3]);
    a = na; b0 = nb0; b1 = nb1; b2 = nb2; b3 = nb3;
  }
  a4[0] = wmma_bf16(a, b0, a4[0]);
  a4[1] = wmma_bf16(a, b1, a4[1]);
  a4[2] = wmma_bf16(a, b2, a4[2]);
  a4[3] = wmma_bf16(a, b3, a4[3]);

  float* dst = acc + (size_t)(bh * 2 + which) * 4096;
#pragma unroll
  for (int j = 0; j < 4; ++j)
#pragma unroll
    for (int r = 0; r < 8; ++r)
      atomicAdd(dst + (wv * 16 + mh + r) * 64 + j * 16 + row, a4[j][r]);
}

__global__ __launch_bounds__(256)
void kvfinish_kernel(const float* __restrict__ acc, const float* __restrict__ b_seq,
                     __bf16* __restrict__ kpT, __bf16* __restrict__ vp) {
  int idx = blockIdx.x * 256 + threadIdx.x;   // 16 * 2 * 4096
  int bh = idx >> 13;
  int rem = idx & 8191;
  int which = rem >> 12;
  int e = rem & 4095;
  int rrow = e >> 6, ccol = e & 63;
  float v = acc[(size_t)(bh * 2 + which) * 4096 + e] +
            b_seq[which == 0 ? rrow : ccol];
  if (which == 0) kpT[(size_t)bh * 4096 + e] = (__bf16)v;
  else            vp [(size_t)bh * 4096 + e] = (__bf16)v;
}

// =====================================================================
// K7: fused attention: attn = softmax(q@kp * temp); o = attn @ vp^T
// wave handles a 16-token strip of one (b,h). 4KB LDS per wave.
// =====================================================================
__global__ __launch_bounds__(256)
void attn_kernel(const __bf16* __restrict__ q, const __bf16* __restrict__ kpT,
                 const __bf16* __restrict__ vp, const float* __restrict__ temp,
                 __bf16* __restrict__ o) {
  __shared__ float sm_all[8 * 16 * 64];
  int wv = threadIdx.x >> 5, lane = threadIdx.x & 31;
  int wid = blockIdx.x * 8 + wv;
  int bh = wid >> 9;
  int rt = wid & 511;
  int b = bh >> 2, h = bh & 3;
  float* sm = sm_all + wv * 1024;
  int row = lane & 15;
  int kb = (lane >> 4) << 3;
  int mh = kb;

  // ---- attn scores: q[16,64] x kp (K = dh = 64) ----
  const __bf16* Ap = q + ((size_t)bh * NNtok + rt * 16 + row) * DHH + kb;
  v8f acc[4] = {v8f{}, v8f{}, v8f{}, v8f{}};
#pragma unroll
  for (int k0 = 0; k0 < 64; k0 += 32) {
    v16bf a = ldfrag(Ap + k0);
#pragma unroll
    for (int j = 0; j < 4; ++j) {
      const __bf16* Bp = kpT + (size_t)bh * 4096 + (j * 16 + row) * 64 + kb + k0;
      acc[j] = wmma_bf16(a, ldfrag(Bp), acc[j]);
    }
  }
#pragma unroll
  for (int j = 0; j < 4; ++j)
#pragma unroll
    for (int r = 0; r < 8; ++r)
      sm[(mh + r) * 64 + j * 16 + row] = acc[j][r];
  __syncthreads();

  // ---- scaled softmax over P=64, one row per lane (lanes 0..15) ----
  if (lane < 16) {
    float tt = temp[h];
    float* rp = sm + lane * 64;
    float mx = -3.4e38f;
    for (int c = 0; c < 64; ++c) mx = fmaxf(mx, rp[c] * tt);
    float s = 0.f;
    for (int c = 0; c < 64; ++c) { float e = __expf(rp[c] * tt - mx); rp[c] = e; s += e; }
    float is = 1.f / s;
    for (int c = 0; c < 64; ++c) rp[c] *= is;
  }
  __syncthreads();

  // ---- o = attn[16,64] x vp^T (K = P = 64) ----
  v8f acc2[4] = {v8f{}, v8f{}, v8f{}, v8f{}};
#pragma unroll
  for (int k0 = 0; k0 < 64; k0 += 32) {
    v16bf a2;
#pragma unroll
    for (int i = 0; i < 8; ++i) {
      a2[i]     = (__bf16)sm[row * 64 + k0 + kb + i];
      a2[i + 8] = (__bf16)sm[row * 64 + k0 + kb + 16 + i];
    }
#pragma unroll
    for (int j = 0; j < 4; ++j) {
      const __bf16* Bp = vp + (size_t)bh * 4096 + (j * 16 + row) * 64 + kb + k0;
      acc2[j] = wmma_bf16(a2, ldfrag(Bp), acc2[j]);
    }
  }
#pragma unroll
  for (int j = 0; j < 4; ++j)
#pragma unroll
    for (int r = 0; r < 8; ++r)
      o[((size_t)b * NNtok + rt * 16 + mh + r) * CC + h * 64 + j * 16 + row] =
          (__bf16)acc2[j][r];
}

// =====================================================================
// K8: LayerNorm over C + residual with x ; one wave per token ->
//     y token-major [B,N,C] f32
// =====================================================================
__global__ __launch_bounds__(256)
void ln_res_kernel(const float* __restrict__ o2, const float* __restrict__ x,
                   const float* __restrict__ g, const float* __restrict__ bta,
                   float* __restrict__ yt) {
  int wv = threadIdx.x >> 5, lane = threadIdx.x & 31;
  int token = blockIdx.x * 8 + wv;
  int b = token >> 13, n = token & (NNtok - 1);
  const float* rowp = o2 + (size_t)token * CC;
  float v[8], s = 0.f, ss = 0.f;
#pragma unroll
  for (int i = 0; i < 8; ++i) {
    v[i] = rowp[lane + 32 * i];
    s += v[i]; ss += v[i] * v[i];
  }
#pragma unroll
  for (int off = 16; off > 0; off >>= 1) {
    s  += __shfl_xor(s, off, 32);
    ss += __shfl_xor(ss, off, 32);
  }
  float mu = s * (1.f / 256.f);
  float var = ss * (1.f / 256.f) - mu * mu;
  float inv = rsqrtf(var + 1e-5f);
  int d = n & 7, w = (n >> 3) & 31, h = n >> 8;
#pragma unroll
  for (int i = 0; i < 8; ++i) {
    int c = lane + 32 * i;
    float nv = (v[i] - mu) * inv * g[c] + bta[c];
    float xv = x[(((size_t)(b * CC + c) * DDm + d) * HHm + h) * WWm + w];
    yt[(size_t)token * CC + c] = xv + nv;
  }
}

// =====================================================================
// K9: depthwise 3x3x3 on token-major y ; threads = channels (coalesced)
// =====================================================================
__global__ __launch_bounds__(256)
void dwconv_kernel(const float* __restrict__ yt, const float* __restrict__ dwk,
                   const float* __restrict__ dwb, __bf16* __restrict__ dwo) {
  int c = threadIdx.x;
  int id = blockIdx.x;                  // B*D*H blocks
  int h = id & 31, d = (id >> 5) & 7, b = id >> 8;
  float wk[27];
#pragma unroll
  for (int t = 0; t < 27; ++t) wk[t] = dwk[c * 27 + t];
  float bias = dwb[c];
  for (int w = 0; w < WWm; ++w) {
    float a = bias;
    for (int kd = 0; kd < 3; ++kd) {
      int d2 = d + kd - 1; if (d2 < 0 || d2 >= DDm) continue;
      for (int kh = 0; kh < 3; ++kh) {
        int h2 = h + kh - 1; if (h2 < 0 || h2 >= HHm) continue;
        for (int kw = 0; kw < 3; ++kw) {
          int w2 = w + kw - 1; if (w2 < 0 || w2 >= WWm) continue;
          int n2 = h2 * 256 + w2 * 8 + d2;
          a += yt[(size_t)(b * NNtok + n2) * CC + c] * wk[kd * 9 + kh * 3 + kw];
        }
      }
    }
    int n = h * 256 + w * 8 + d;
    dwo[(size_t)(b * NNtok + n) * CC + c] = (__bf16)a;
  }
}

// =====================================================================
// K10: final residual back to NCDHW
// =====================================================================
__global__ __launch_bounds__(256)
void final_kernel(const float* __restrict__ yt, const float* __restrict__ pw,
                  float* __restrict__ out) {
  int idx = blockIdx.x * 256 + threadIdx.x;
  int w = idx & 31, h = (idx >> 5) & 31, d = (idx >> 10) & 7;
  int c = (idx >> 13) & 255, b = idx >> 21;
  int n = h * 256 + w * 8 + d;
  size_t tok = (size_t)(b * NNtok + n) * CC + c;
  out[idx] = yt[tok] + pw[tok];
}

// =====================================================================
extern "C" void kernel_launch(void* const* d_in, const int* in_sizes, int n_in,
                              void* d_out, int out_size, void* d_ws, size_t ws_size,
                              hipStream_t stream) {
  const float* x      = (const float*)d_in[0];
  const float* w_qkv  = (const float*)d_in[1];
  const float* w_seq  = (const float*)d_in[2];
  const float* b_seq  = (const float*)d_in[3];
  const float* temp   = (const float*)d_in[4];
  const float* w_out  = (const float*)d_in[5];
  const float* b_out  = (const float*)d_in[6];
  const float* ln_g   = (const float*)d_in[7];
  const float* ln_b   = (const float*)d_in[8];
  const float* dw_k   = (const float*)d_in[9];
  const float* dw_b   = (const float*)d_in[10];
  const float* pw_k   = (const float*)d_in[11];
  const float* pw_b   = (const float*)d_in[12];
  float* out = (float*)d_out;
  char* ws = (char*)d_ws;

  // ---- workspace layout (bytes), with lifetime-based aliasing ----
  const size_t OFF_QKV   = 0;                         // f32 [32768,768] = 100.66MB
  const size_t OFF_O2    = 0;                         // f32 [32768,256] (after qkv dead)
  const size_t OFF_PW    = 33554432;                  // f32 [32768,256] (inside qkv region)
  const size_t OFF_T     = 100663296;                 // bf16 [32768,256] = 16.78MB
  const size_t OFF_O     = OFF_T;                     // bf16 o (after t dead)
  const size_t OFF_Q     = 117440512;                 // bf16 q (later dw output)
  const size_t OFF_DW    = OFF_Q;
  const size_t OFF_KT    = 134217728;                 // bf16 kT 16.78MB
  const size_t OFF_VT    = 150994944;                 // bf16 vT 16.78MB
  const size_t OFF_YT    = OFF_KT;                    // f32 y token-major 33.55MB (after kT/vT dead)
  const size_t OFF_WQKVT = 167772160;                 // bf16 [768,256]
  const size_t OFF_WOUTT = 168165376;                 // bf16 [256,256]
  const size_t OFF_WSEQT = 168296448;                 // bf16 [64,8192]
  const size_t OFF_PWT   = 169345024;                 // bf16 [256,256]
  const size_t OFF_NORM  = 169476096;                 // f32 [4,512]
  const size_t OFF_KVACC = 169484288;                 // f32 131072
  const size_t OFF_KPT   = 170008576;                 // bf16 16*4096
  const size_t OFF_VP    = 170139648;                 // bf16 16*4096

  float*  qkv   = (float*)(ws + OFF_QKV);
  float*  o2    = (float*)(ws + OFF_O2);
  float*  pw    = (float*)(ws + OFF_PW);
  __bf16* t     = (__bf16*)(ws + OFF_T);
  __bf16* obuf  = (__bf16*)(ws + OFF_O);
  __bf16* qb    = (__bf16*)(ws + OFF_Q);
  __bf16* dwo   = (__bf16*)(ws + OFF_DW);
  __bf16* kT    = (__bf16*)(ws + OFF_KT);
  __bf16* vT    = (__bf16*)(ws + OFF_VT);
  float*  yt    = (float*)(ws + OFF_YT);
  __bf16* wqkvT = (__bf16*)(ws + OFF_WQKVT);
  __bf16* woutT = (__bf16*)(ws + OFF_WOUTT);
  __bf16* wseqT = (__bf16*)(ws + OFF_WSEQT);
  __bf16* pwT   = (__bf16*)(ws + OFF_PWT);
  float*  norms = (float*)(ws + OFF_NORM);
  float*  kvacc = (float*)(ws + OFF_KVACC);
  __bf16* kpT   = (__bf16*)(ws + OFF_KPT);
  __bf16* vp    = (__bf16*)(ws + OFF_VP);

  // 1. pack weights + zero scratch
  pack_weights_kernel<<<3849, 256, 0, stream>>>(w_qkv, w_out, w_seq, pw_k,
      wqkvT, woutT, wseqT, pwT, norms, kvacc);
  // 2. x -> token-major bf16
  pack_x_kernel<<<MTOT * CC / 256, 256, 0, stream>>>(x, t);
  // 3. qkv GEMM: [32768,256] x [256,768]  (32-row wave tiles)
  gemm_bf16_kernel<<<(MTOT / 32) * (768 / 64) / 8, 256, 0, stream>>>(
      t, wqkvT, nullptr, qkv, MTOT, 768, 256);
  // 4. token-axis L2 norms for q,k
  colnorm_kernel<<<BB * (NNtok / 64), 256, 0, stream>>>(qkv, norms);
  norm_finish_kernel<<<8, 256, 0, stream>>>(norms);
  // 5. split + normalize + transpose
  pack_qkv_kernel<<<MTOT * CC / 256, 256, 0, stream>>>(qkv, norms, qb, kT, vT);
  // 6. sequence projections (split-K WMMA + atomic reduce), then +bias -> bf16
  kvproj_kernel<<<BB * NHH * 32 * 2, 128, 0, stream>>>(wseqT, kT, vT, kvacc);
  kvfinish_kernel<<<131072 / 256, 256, 0, stream>>>(kvacc, b_seq, kpT, vp);
  // 7. fused attention (scores -> softmax -> output)
  attn_kernel<<<BB * NHH * (NNtok / 16) / 8, 256, 0, stream>>>(
      qb, kpT, vp, temp, obuf);
  // 8. out-proj GEMM + LayerNorm + residual
  gemm_bf16_kernel<<<(MTOT / 32) * (256 / 64) / 8, 256, 0, stream>>>(
      obuf, woutT, b_out, o2, MTOT, 256, 256);
  ln_res_kernel<<<MTOT / 8, 256, 0, stream>>>(o2, x, ln_g, ln_b, yt);
  // 9. DSConv: depthwise 3x3x3, then pointwise GEMM
  dwconv_kernel<<<BB * DDm * HHm, 256, 0, stream>>>(yt, dw_k, dw_b, dwo);
  gemm_bf16_kernel<<<(MTOT / 32) * (256 / 64) / 8, 256, 0, stream>>>(
      dwo, pwT, pw_b, pw, MTOT, 256, 256);
  // 10. final residual back to NCDHW
  final_kernel<<<MTOT * CC / 256, 256, 0, stream>>>(yt, pw, out);
}
